// AttentionPI_36129264894201
// MI455X (gfx1250) — matrix operand order repaired and checked by reading, not compiled, not run on hardware
//
#include <hip/hip_runtime.h>
#include <hip/hip_bf16.h>

// ---------------------------------------------------------------------------
// AttentionPI for MI455X (gfx1250): fused Gaussian-attention + 1x1 convs.
// fp32 VALU softmax (attns is a required fp32 output); bf16 WMMA
// (v_wmma_f32_16x16x32_bf16, fp32 accumulate) for all three GEMMs.
// Weights converted fp32->bf16 once (kernel 0). GEMM loops are register-
// blocked (4 m-tiles / wave) AND software-pipelined with double-buffered
// fragment registers (prefetch distance 1-2) so WMMAs never wait on loads
// issued in the same k-step.
// ---------------------------------------------------------------------------

#define NB   16     // batch
#define NC   512    // channels C
#define N2C  1024   // 2C
#define NH   8      // heads
#define ND   128    // 2C / H
#define NTPI 2048   // mel frames (queries)
#define NTP  512    // text tokens (keys)

typedef __attribute__((ext_vector_type(16))) __bf16 v16bf;
typedef __attribute__((ext_vector_type(8)))  __bf16 v8bf;
typedef __attribute__((ext_vector_type(4)))  __bf16 v4bf;
typedef __attribute__((ext_vector_type(8)))  float  v8f;
typedef __attribute__((ext_vector_type(4)))  float  v4f;

__device__ __forceinline__ v16bf cat16(v8bf lo, v8bf hi) {
    return __builtin_shufflevector(lo, hi, 0,1,2,3,4,5,6,7,8,9,10,11,12,13,14,15);
}

__device__ __forceinline__ v8f wmma_bf16(v16bf a, v16bf b, v8f c) {
    // D = A(16x32 bf16) * B(32x16 bf16) + C(16x16 f32)
    return __builtin_amdgcn_wmma_f32_16x16x32_bf16(
        /*neg_a=*/false, a, /*neg_b=*/false, b,
        /*c_mod=*/(short)0, c, /*reuse_a=*/false, /*reuse_b=*/false);
}

__device__ __forceinline__ v16bf ldg16(const __bf16* p) {
    v8bf lo = *(const v8bf*)(p);
    v8bf hi = *(const v8bf*)(p + 8);        // 16 contiguous bf16
    return cat16(lo, hi);
}

__device__ __forceinline__ v16bf lds16c(const __bf16* p) {
    v8bf lo = *(const v8bf*)(p);
    v8bf hi = *(const v8bf*)(p + 8);        // 16 contiguous bf16
    return cat16(lo, hi);
}

// ---------------------------------------------------------------------------
// Kernel 0: one-shot fp32->bf16 conversion of w_hidden and w_out (+ sigma).
// ---------------------------------------------------------------------------
__global__ __launch_bounds__(256)
void k0_convert(const float* __restrict__ wh,     // (2C, C)   = 524288
                const float* __restrict__ wout,   // (C, 2C)   = 524288
                const float* __restrict__ sigma,  // (H,)
                __bf16* __restrict__ wh_bf,
                __bf16* __restrict__ wout_bf,
                float*  __restrict__ sigma_out)
{
    const size_t idx = ((size_t)blockIdx.x * 256 + threadIdx.x) * 4;
    {
        v4f a = *(const v4f*)(wh + idx);
        v4bf o; o[0]=(__bf16)a[0]; o[1]=(__bf16)a[1]; o[2]=(__bf16)a[2]; o[3]=(__bf16)a[3];
        *(v4bf*)(wh_bf + idx) = o;
    }
    {
        v4f a = *(const v4f*)(wout + idx);
        v4bf o; o[0]=(__bf16)a[0]; o[1]=(__bf16)a[1]; o[2]=(__bf16)a[2]; o[3]=(__bf16)a[3];
        *(v4bf*)(wout_bf + idx) = o;
    }
    if (blockIdx.x == 0 && threadIdx.x < NH) {
        float s = sigma[threadIdx.x];
        sigma_out[threadIdx.x] = fminf(fmaxf(s, 1e-6f), 3.0f);
    }
}

// ---------------------------------------------------------------------------
// Kernel 1: h = w_hidden @ x_h + b_hidden  (per batch: M=1024, N=512, K=512)
// Output bf16, t-contiguous: hT[b][o][t]  (o = h*128 + dd).
// Grid: 512 blocks (b x t-tile), 256 threads (8 waves).
// Wave: 2 passes of 4 m-tiles; fragments double-buffered, prefetch dist 1.
// ---------------------------------------------------------------------------
__global__ __launch_bounds__(256)
void k1_hidden(const float*  __restrict__ xh,     // (B, C, T_P)
               const __bf16* __restrict__ wh_bf,  // (2C, C) bf16
               const float*  __restrict__ bh,     // (2C,)
               __bf16* __restrict__ hT)           // ws: (B, 2C, T_P) bf16
{
    __shared__ __bf16 xT[16 * (NC + 8)];   // x tile transposed [t][c], row-padded
    __shared__ float  bhid[N2C];

    const int tid = threadIdx.x;
    const int b   = blockIdx.x >> 5;         // 32 t-tiles per batch
    const int t0  = (blockIdx.x & 31) * 16;

    // Stage x_h[:, t0:t0+16] transposed -> LDS bf16 (coalesced 64B rows in).
    for (int j = 0; j < 32; ++j) {
        int idx = tid + 256 * j;             // 8192 elems
        int c = idx >> 4, n = idx & 15;
        xT[n * (NC + 8) + c] = (__bf16)xh[((size_t)b * NC + c) * NTP + t0 + n];
    }
    for (int j = tid; j < N2C; j += 256) bhid[j] = bh[j];
    __syncthreads();

    const int wv = tid >> 5;
    const int lane = tid & 31;
    const int lg = lane >> 4;      // lane group (ISA 16-bit A/B layouts)
    const int lm = lane & 15;

    const __bf16* xTl = xT + lm * (NC + 8) + 16 * lg;

    for (int half = 0; half < 2; ++half) {
        const int o0 = wv * 128 + half * 64;     // 4 m-tiles: o0 + 16*mt
        const __bf16* whl = wh_bf + (size_t)(o0 + lm) * NC + 16 * lg;
        v8f acc[4] = {};
        v16bf Ab[2][4];
        v16bf Bb[2];
        // prologue: ks = 0
        Bb[0] = lds16c(xTl);
#pragma unroll
        for (int mt = 0; mt < 4; ++mt) Ab[0][mt] = ldg16(whl + (size_t)(16 * mt) * NC);
#pragma unroll
        for (int ks = 0; ks < 16; ++ks) {
            const int cb = ks & 1, nb = cb ^ 1;
            if (ks < 15) {                        // prefetch ks+1 (dist 1)
                const int c1 = (ks + 1) * 32;
                Bb[nb] = lds16c(xTl + c1);
#pragma unroll
                for (int mt = 0; mt < 4; ++mt)
                    Ab[nb][mt] = ldg16(whl + (size_t)(16 * mt) * NC + c1);
            }
#pragma unroll
            for (int mt = 0; mt < 4; ++mt)
                acc[mt] = wmma_bf16(Ab[cb][mt], Bb[cb], acc[mt]);
        }
        // D(m,n): lane = n + 16*(m>>3), vgpr r -> m = r + 8*lg, n = lm.
#pragma unroll
        for (int mt = 0; mt < 4; ++mt) {
#pragma unroll
            for (int r = 0; r < 8; ++r) {
                int o = o0 + 16 * mt + r + 8 * lg;
                hT[((size_t)b * N2C + o) * NTP + t0 + lm] = (__bf16)(acc[mt][r] + bhid[o]);
            }
        }
    }
}

// ---------------------------------------------------------------------------
// Kernel 2: per (b, 16-query tile): softmax (fp32) -> attns out + bf16 probs,
// out1 = attns @ h (WMMA, per head), fused out = w_out @ out1 + b_out (WMMA).
// Grid: 2048 blocks (b x 128 q-tiles), 256 threads (8 waves).
// ---------------------------------------------------------------------------
__global__ __launch_bounds__(256)
void k2_attn(const float* __restrict__ pi,       // (B, T_PI)
             const float* __restrict__ p,        // (B, T_P)
             const unsigned char* __restrict__ mask, // (B, T_P) bool
             const float* __restrict__ sigma,    // (H,)
             const float* __restrict__ bout_g,   // (C,)
             const __bf16* __restrict__ hT,      // (B, 2C, T_P) bf16
             const __bf16* __restrict__ wout_bf, // (C, 2C) bf16
             float* __restrict__ out,            // (B, C, T_PI)
             float* __restrict__ attns)          // (B, H, T_PI, T_P)
{
    __shared__ __bf16 probs[16 * (NTP + 8)];   // [q][t], row-padded
    __shared__ __bf16 out1 [16 * (N2C + 8)];   // [q][o2], row-padded
    __shared__ float  pvals[NTP];
    __shared__ float  maskadd[NTP];            // 0 or -inf
    __shared__ float  piv[16];
    __shared__ float  sigS[NH];
    __shared__ float  boutS[NC];

    const int tid = threadIdx.x;
    const int b   = blockIdx.x >> 7;           // 128 q-tiles per batch
    const int q0  = (blockIdx.x & 127) * 16;
    const int wv = tid >> 5;
    const int lane = tid & 31;
    const int lg = lane >> 4;
    const int lm = lane & 15;
    const float NEG_INF = -__builtin_inff();

    for (int j = tid; j < NTP; j += 256) {
        pvals[j]   = p[(size_t)b * NTP + j];
        maskadd[j] = mask[(size_t)b * NTP + j] ? 0.0f : NEG_INF;
    }
    for (int j = tid; j < NC; j += 256) boutS[j] = bout_g[j];
    if (tid < 16) piv[tid] = pi[(size_t)b * NTPI + q0 + tid];
    if (tid < NH) sigS[tid] = fminf(fmaxf(sigma[tid], 1e-6f), 3.0f);
    __syncthreads();

    for (int h = 0; h < NH; ++h) {
        const float sg = sigS[h];
        // ---- softmax: wave wv owns query rows {2wv, 2wv+1}, 512 keys ----
        for (int rq = 0; rq < 2; ++rq) {
            const int q = wv * 2 + rq;
            const float piq = piv[q];
            float e[16];
            float mx = NEG_INF;
#pragma unroll
            for (int i = 0; i < 16; ++i) {
                int t = lane + 32 * i;
                float d = piq - pvals[t];
                float v = -(d * d) * sg + maskadd[t];
                e[i] = v;
                mx = fmaxf(mx, v);
            }
            for (int off = 16; off; off >>= 1)
                mx = fmaxf(mx, __shfl_xor(mx, off, 32));
            float sum = 0.0f;
#pragma unroll
            for (int i = 0; i < 16; ++i) {
                float ex = __expf(e[i] - mx);
                e[i] = ex;
                sum += ex;
            }
            for (int off = 16; off; off >>= 1)
                sum += __shfl_xor(sum, off, 32);
            const float inv = 1.0f / sum;
            float* arow = attns + ((size_t)(b * NH + h) * NTPI + q0 + q) * NTP;
#pragma unroll
            for (int i = 0; i < 16; ++i) {
                int t = lane + 32 * i;
                float pr = e[i] * inv;
                arow[t] = pr;                      // fp32 required output
                probs[q * (NTP + 8) + t] = (__bf16)pr;
            }
        }
        __syncthreads();

        // ---- out1[q][h*128+dd] = probs(16x512) @ hT(512x128) ----
        // Wave owns one 16-wide d-tile; 2 accumulators (even/odd k-steps)
        // break the WMMA chain; 3-deep fragment ring, prefetch distance 2.
        {
            const int dd0 = wv * 16;
            const __bf16* hl = hT + (size_t)(b * NH + h) * ND * NTP
                                  + (size_t)(dd0 + lm) * NTP + 16 * lg;
            const __bf16* pl = probs + lm * (NTP + 8) + 16 * lg;
            v8f accE = {}, accO = {};
            v16bf Ab[3], Bb[3];
            Ab[0] = ldg16(pl);          Bb[0] = lds16c(hl);
            Ab[1] = ldg16(pl + 32);     Bb[1] = lds16c(hl + 32);
#pragma unroll
            for (int ks = 0; ks < 16; ++ks) {
                const int cb = ks % 3;
                if (ks < 14) {                    // prefetch ks+2 (dist 2)
                    const int pb = (ks + 2) % 3;
                    const int k2 = (ks + 2) * 32;
                    Ab[pb] = ldg16(pl + k2);
                    Bb[pb] = lds16c(hl + k2);
                }
                if (ks & 1) accO = wmma_bf16(Ab[cb], Bb[cb], accO);
                else        accE = wmma_bf16(Ab[cb], Bb[cb], accE);
            }
            v8f acc = accE + accO;
#pragma unroll
            for (int r = 0; r < 8; ++r) {
                int q  = r + 8 * lg;
                int o2 = h * ND + dd0 + lm;
                out1[q * (N2C + 8) + o2] = (__bf16)acc[r];
            }
        }
        __syncthreads();   // protect probs reuse next head + out1 publication
    }

    // ---- out[:, q0:q0+16] = w_out(512x1024) @ out1(1024x16) + b_out ----
    // Wave owns 64 output rows as 4 m-tiles; shared B fragment, double-
    // buffered fragments, prefetch distance 1.
    {
        const int m0 = wv * 64;
        const __bf16* wl = wout_bf + (size_t)(m0 + lm) * N2C + 16 * lg;
        const __bf16* ol = out1 + lm * (N2C + 8) + 16 * lg;
        v8f acc[4] = {};
        v16bf Ab[2][4];
        v16bf Bb[2];
        Bb[0] = lds16c(ol);
#pragma unroll
        for (int mt = 0; mt < 4; ++mt) Ab[0][mt] = ldg16(wl + (size_t)(16 * mt) * N2C);
#pragma unroll
        for (int ks = 0; ks < 32; ++ks) {
            const int cb = ks & 1, nb = cb ^ 1;
            if (ks < 31) {                        // prefetch ks+1 (dist 1)
                const int k1 = (ks + 1) * 32;
                Bb[nb] = lds16c(ol + k1);
#pragma unroll
                for (int mt = 0; mt < 4; ++mt)
                    Ab[nb][mt] = ldg16(wl + (size_t)(16 * mt) * N2C + k1);
            }
#pragma unroll
            for (int mt = 0; mt < 4; ++mt)
                acc[mt] = wmma_bf16(Ab[cb][mt], Bb[cb], acc[mt]);
        }
#pragma unroll
        for (int mt = 0; mt < 4; ++mt) {
#pragma unroll
            for (int r = 0; r < 8; ++r) {
                int oc = m0 + 16 * mt + r + 8 * lg;
                out[((size_t)b * NC + oc) * NTPI + q0 + lm] = acc[mt][r] + boutS[oc];
            }
        }
    }
}

// ---------------------------------------------------------------------------
extern "C" void kernel_launch(void* const* d_in, const int* in_sizes, int n_in,
                              void* d_out, int out_size, void* d_ws, size_t ws_size,
                              hipStream_t stream) {
    const float*         pi_   = (const float*)d_in[0];
    const float*         p_    = (const float*)d_in[1];
    const float*         xh    = (const float*)d_in[2];
    const unsigned char* mask  = (const unsigned char*)d_in[3];
    const float*         sigma = (const float*)d_in[4];
    const float*         wh    = (const float*)d_in[5];
    const float*         bh    = (const float*)d_in[6];
    const float*         wout  = (const float*)d_in[7];
    const float*         bout  = (const float*)d_in[8];

    float* out     = (float*)d_out;                                   // (B,C,T_PI)
    float* attns   = out + (size_t)NB * NC * NTPI;                    // (B,H,T_PI,T_P)
    float* sig_out = attns + (size_t)NB * NH * NTPI * NTP;            // (H,)

    __bf16* hT      = (__bf16*)d_ws;                                  // 16 MB
    __bf16* wout_bf = hT + (size_t)NB * N2C * NTP;                    // +1 MB
    __bf16* wh_bf   = wout_bf + (size_t)NC * N2C;                     // +1 MB

    k0_convert<<<512, 256, 0, stream>>>(wh, wout, sigma, wh_bf, wout_bf, sig_out);
    k1_hidden<<<NB * (NTP / 16), 256, 0, stream>>>(xh, wh_bf, bh, hT);
    k2_attn<<<NB * (NTPI / 16), 256, 0, stream>>>(
        pi_, p_, mask, sigma, bout, hT, wout_bf, out, attns);
}
